// Phi4VisionSiglip2VisionTransformer_41626823033117
// MI455X (gfx1250) — compile-verified, matrix operand order checked by reading, not run on hardware
//
#include <hip/hip_runtime.h>
#include <hip/hip_bf16.h>
#include <math.h>

typedef __attribute__((ext_vector_type(16))) __bf16 v16bf;
typedef __attribute__((ext_vector_type(8)))  __bf16 v8bf;
typedef __attribute__((ext_vector_type(8)))  float  v8f;

#define WG 256
#define BM 128
#define BN 128
#define BK 32
#define LDT (BK + 8)   // 40 bf16 = 80 B, multiple of 16 B
#define TQ 32
#define TK 64
#define KDP 96
#define LDP (TK + 8)   // 72 bf16, multiple of 8

__device__ inline v8f vzero8() {
  v8f z;
#pragma unroll
  for (int i = 0; i < 8; ++i) z[i] = 0.f;
  return z;
}

__device__ inline float gelu_tanh(float x) {
  const float c0 = 0.7978845608028654f;
  return 0.5f * x * (1.f + tanhf(c0 * (x + 0.044715f * x * x * x)));
}

// A-style 16x32 bf16 fragment, wide-load version.
// Lane L: row = row0 + (L&15), ko = k0 + (L>>4)*8.
// f[0..7] = row[ko..ko+7], f[8..15] = row[ko+16..ko+23]  -> two b128 LDS loads.
// Requires: ld multiple of 8 bf16, k0 multiple of 32, base 16B-aligned.
__device__ inline v16bf frag_ldsx2(const __bf16* p, int ld, int row0, int k0, int lane) {
  const __bf16* rp = p + (size_t)(row0 + (lane & 15)) * ld + k0 + ((lane >> 4) << 3);
  v8bf lo = *reinterpret_cast<const v8bf*>(rp);
  v8bf hi = *reinterpret_cast<const v8bf*>(rp + 16);
  return __builtin_shufflevector(lo, hi, 0, 1, 2, 3, 4, 5, 6, 7,
                                 8, 9, 10, 11, 12, 13, 14, 15);
}

// ---------------------------------------------------------------------------
// Tiled WMMA GEMM: C[M,N] = act(A[M,K] @ W[K,N] + bias) + resid
// 128x128 block tile, 8 waves x (32x64) subtiles. Both A and B tiles stored
// K-contiguous in LDS (B transposed) so every fragment is 2x ds_load_b128.
// ---------------------------------------------------------------------------
__global__ __launch_bounds__(WG) void gemm_bf16(
    const float* __restrict__ A, const float* __restrict__ W,
    const float* __restrict__ bias, const float* __restrict__ resid,
    float* __restrict__ C, int M, int N, int K, int act) {
  __shared__ __align__(16) __bf16 sA[BM][LDT];
  __shared__ __align__(16) __bf16 sBt[BN][LDT];  // [n][k]
  const int tid = threadIdx.x, lane = tid & 31, wv = tid >> 5;
  const int bm = blockIdx.y * BM, bn = blockIdx.x * BN;
  const int wm = (wv & 3) * 32, wn = (wv >> 2) * 64;
  v8f acc[2][4];
#pragma unroll
  for (int i = 0; i < 2; ++i)
#pragma unroll
    for (int j = 0; j < 4; ++j) acc[i][j] = vzero8();

  for (int k0 = 0; k0 < K; k0 += BK) {
    // A tile: 128x32, float4 loads (K is always a multiple of 4).
#pragma unroll
    for (int i = tid; i < (BM * BK) / 4; i += WG) {
      int r = i >> 3, c = (i & 7) << 2;
      int gm = bm + r, gk = k0 + c;
      float4 f4 = make_float4(0.f, 0.f, 0.f, 0.f);
      if (gm < M && gk < K)
        f4 = *reinterpret_cast<const float4*>(&A[(size_t)gm * K + gk]);
      sA[r][c + 0] = (__bf16)f4.x;
      sA[r][c + 1] = (__bf16)f4.y;
      sA[r][c + 2] = (__bf16)f4.z;
      sA[r][c + 3] = (__bf16)f4.w;
    }
    // B tile: 32x128 from [K][N], stored transposed [n][k].
#pragma unroll
    for (int i = tid; i < (BK * BN) / 4; i += WG) {
      int r = i >> 5, c = (i & 31) << 2;
      int gk = k0 + r, gn = bn + c;
      float4 f4 = make_float4(0.f, 0.f, 0.f, 0.f);
      if (gk < K && gn < N)
        f4 = *reinterpret_cast<const float4*>(&W[(size_t)gk * N + gn]);
      sBt[c + 0][r] = (__bf16)f4.x;
      sBt[c + 1][r] = (__bf16)f4.y;
      sBt[c + 2][r] = (__bf16)f4.z;
      sBt[c + 3][r] = (__bf16)f4.w;
    }
    // prefetch next K tile into caches (global_prefetch_b8)
    if (k0 + BK < K) {
      int gm = bm + (tid >> 1);
      if (gm < M) __builtin_prefetch(&A[(size_t)gm * K + k0 + BK + (tid & 1) * 16], 0, 1);
      int gn = bn + (tid >> 1);
      if (gn < N) __builtin_prefetch(&W[(size_t)(k0 + BK + (tid & 1) * 16) * N + gn], 0, 1);
    }
    __syncthreads();
    v16bf af[2], bw[4];
#pragma unroll
    for (int mt = 0; mt < 2; ++mt)
      af[mt] = frag_ldsx2(&sA[0][0], LDT, wm + mt * 16, 0, lane);
#pragma unroll
    for (int nt = 0; nt < 4; ++nt)
      bw[nt] = frag_ldsx2(&sBt[0][0], LDT, wn + nt * 16, 0, lane);
#pragma unroll
    for (int mt = 0; mt < 2; ++mt)
#pragma unroll
      for (int nt = 0; nt < 4; ++nt)
        acc[mt][nt] = __builtin_amdgcn_wmma_f32_16x16x32_bf16(
            false, af[mt], false, bw[nt], (short)0, acc[mt][nt], false, false);
    __syncthreads();
  }
#pragma unroll
  for (int mt = 0; mt < 2; ++mt)
#pragma unroll
    for (int nt = 0; nt < 4; ++nt)
#pragma unroll
      for (int r = 0; r < 8; ++r) {
        int m = bm + wm + mt * 16 + ((lane >> 4) << 3) + r;
        int n = bn + wn + nt * 16 + (lane & 15);
        if (m < M && n < N) {
          float v = acc[mt][nt][r];
          if (bias)  v += bias[n];
          if (act)   v = gelu_tanh(v);
          if (resid) v += resid[(size_t)m * N + n];
          C[(size_t)m * N + n] = v;
        }
      }
}

// ---------------------------------------------------------------------------
// Flash attention: one block per (b, head, 32-query tile), streaming softmax.
// Head dim 72 zero-padded to 96; QK^T and P@V both WMMA bf16.
// V stored transposed and P stored as bf16 so all fragments are wide loads.
// ---------------------------------------------------------------------------
__global__ __launch_bounds__(WG) void flash_attn(
    const float* __restrict__ Q, const float* __restrict__ Kx,
    const float* __restrict__ Vx, const float* __restrict__ mask,
    float* __restrict__ Oo, int S, int NH, int HDv, float scale) {
  __shared__ __align__(16) __bf16 sQ[TQ][KDP];
  __shared__ __align__(16) __bf16 sK[TK][KDP];
  __shared__ __align__(16) __bf16 sVt[KDP][TK];  // [d][key]
  __shared__ __align__(16) __bf16 sP[TQ][LDP];
  __shared__ float sS[TQ][TK];
  __shared__ float sO[TQ][KDP];
  __shared__ float sM[TQ], sL[TQ], sAl[TQ];
  const int tid = threadIdx.x, lane = tid & 31, wv = tid >> 5;
  const int b = blockIdx.z, hh = blockIdx.y, q0 = blockIdx.x * TQ;
  const int D = NH * HDv;

#pragma unroll
  for (int i = tid; i < TQ * KDP / 4; i += WG) {
    int r = i / (KDP / 4), c = (i % (KDP / 4)) * 4;
    float4 f4 = make_float4(0.f, 0.f, 0.f, 0.f);
    if (c < HDv)
      f4 = *reinterpret_cast<const float4*>(&Q[((size_t)b * S + q0 + r) * D + hh * HDv + c]);
    sQ[r][c + 0] = (__bf16)f4.x;
    sQ[r][c + 1] = (__bf16)f4.y;
    sQ[r][c + 2] = (__bf16)f4.z;
    sQ[r][c + 3] = (__bf16)f4.w;
    sO[r][c + 0] = 0.f; sO[r][c + 1] = 0.f; sO[r][c + 2] = 0.f; sO[r][c + 3] = 0.f;
  }
  if (tid < TQ) { sM[tid] = -1.0e30f; sL[tid] = 0.f; }
  __syncthreads();

  for (int kt = 0; kt < S; kt += TK) {
#pragma unroll
    for (int i = tid; i < TK * KDP / 4; i += WG) {
      int r = i / (KDP / 4), c = (i % (KDP / 4)) * 4;
      float4 kv = make_float4(0.f, 0.f, 0.f, 0.f);
      float4 vv = make_float4(0.f, 0.f, 0.f, 0.f);
      if (c < HDv) {
        size_t gi = ((size_t)b * S + kt + r) * D + hh * HDv + c;
        kv = *reinterpret_cast<const float4*>(&Kx[gi]);
        vv = *reinterpret_cast<const float4*>(&Vx[gi]);
      }
      sK[r][c + 0] = (__bf16)kv.x;
      sK[r][c + 1] = (__bf16)kv.y;
      sK[r][c + 2] = (__bf16)kv.z;
      sK[r][c + 3] = (__bf16)kv.w;
      sVt[c + 0][r] = (__bf16)vv.x;
      sVt[c + 1][r] = (__bf16)vv.y;
      sVt[c + 2][r] = (__bf16)vv.z;
      sVt[c + 3][r] = (__bf16)vv.w;
    }
    __syncthreads();
    // S_tile = Q @ K^T : 32x64 output = 8 WMMA tiles, one per wave.
    {
      int mt = (wv & 1) * 16, nt = (wv >> 1) * 16;
      v8f acc = vzero8();
#pragma unroll
      for (int kk = 0; kk < KDP; kk += 32) {
        v16bf a  = frag_ldsx2(&sQ[0][0], KDP, mt, kk, lane);
        v16bf bb = frag_ldsx2(&sK[0][0], KDP, nt, kk, lane); // K^T: (n=key,k=d)
        acc = __builtin_amdgcn_wmma_f32_16x16x32_bf16(false, a, false, bb,
                                                      (short)0, acc, false, false);
      }
#pragma unroll
      for (int r = 0; r < 8; ++r) {
        int m = mt + ((lane >> 4) << 3) + r;
        int n = nt + (lane & 15);
        float mk = (1.f - mask[(size_t)b * S + kt + n]) * -3.4028235e38f;
        sS[m][n] = acc[r] * scale + mk;
      }
    }
    __syncthreads();
    // streaming softmax per row (one wave handles the 32 rows); write P in bf16.
    if (tid < TQ) {
      float mo = sM[tid], mn = mo;
      for (int c = 0; c < TK; ++c) mn = fmaxf(mn, sS[tid][c]);
      float al = __expf(mo - mn);
      float ls = 0.f;
      for (int c = 0; c < TK; ++c) {
        float p = __expf(sS[tid][c] - mn);
        sP[tid][c] = (__bf16)p;
        ls += p;
      }
      sL[tid] = sL[tid] * al + ls;
      sM[tid] = mn;
      sAl[tid] = al;
    }
    __syncthreads();
    // O = O*alpha + P @ V : 32x96 output = 12 WMMA tiles over 8 waves.
    for (int t = wv; t < 12; t += 8) {
      int mt = (t & 1) * 16, nt = (t >> 1) * 16;
      v8f acc = vzero8();
#pragma unroll
      for (int kk = 0; kk < TK; kk += 32) {
        v16bf a  = frag_ldsx2(&sP[0][0], LDP, mt, kk, lane);
        v16bf bb = frag_ldsx2(&sVt[0][0], TK, nt, kk, lane); // V^T: (n=d,k=key)
        acc = __builtin_amdgcn_wmma_f32_16x16x32_bf16(false, a, false, bb,
                                                      (short)0, acc, false, false);
      }
#pragma unroll
      for (int r = 0; r < 8; ++r) {
        int m = mt + ((lane >> 4) << 3) + r;
        int n = nt + (lane & 15);
        sO[m][n] = sO[m][n] * sAl[m] + acc[r];
      }
    }
    __syncthreads();
  }
#pragma unroll
  for (int i = tid; i < TQ * HDv / 4; i += WG) {
    int r = i / (HDv / 4), c = (i % (HDv / 4)) * 4;
    float inv = 1.f / sL[r];
    float4 o;
    o.x = sO[r][c + 0] * inv;
    o.y = sO[r][c + 1] * inv;
    o.z = sO[r][c + 2] * inv;
    o.w = sO[r][c + 3] * inv;
    *reinterpret_cast<float4*>(&Oo[((size_t)b * S + q0 + r) * D + hh * HDv + c]) = o;
  }
}

// ---------------------------------------------------------------------------
// LayerNorm: one block per token.
// ---------------------------------------------------------------------------
__global__ __launch_bounds__(WG) void ln_fwd(const float* __restrict__ X,
                                             const float* __restrict__ g,
                                             const float* __restrict__ bb,
                                             float* __restrict__ Y, int Dv) {
  const float* x = X + (size_t)blockIdx.x * Dv;
  float* y = Y + (size_t)blockIdx.x * Dv;
  __shared__ float red[WG];
  float s = 0.f;
  for (int d = threadIdx.x; d < Dv; d += WG) s += x[d];
  red[threadIdx.x] = s;
  __syncthreads();
  for (int st = WG / 2; st > 0; st >>= 1) {
    if (threadIdx.x < st) red[threadIdx.x] += red[threadIdx.x + st];
    __syncthreads();
  }
  float mean = red[0] / (float)Dv;
  __syncthreads();
  float s2 = 0.f;
  for (int d = threadIdx.x; d < Dv; d += WG) {
    float t = x[d] - mean;
    s2 += t * t;
  }
  red[threadIdx.x] = s2;
  __syncthreads();
  for (int st = WG / 2; st > 0; st >>= 1) {
    if (threadIdx.x < st) red[threadIdx.x] += red[threadIdx.x + st];
    __syncthreads();
  }
  float inv = rsqrtf(red[0] / (float)Dv + 1e-6f);
  for (int d = threadIdx.x; d < Dv; d += WG)
    y[d] = (x[d] - mean) * inv * g[d] + bb[d];
}

// ---------------------------------------------------------------------------
// Bilinear (half-pixel) resize of 16x16 pos-embed to (h,w), added to hidden.
// ---------------------------------------------------------------------------
__global__ void posemb_add(float* __restrict__ hbuf, const float* __restrict__ pe,
                           const int* __restrict__ ss, int S, int Dv, int PESv) {
  int b = blockIdx.y;
  int idx = blockIdx.x * blockDim.x + threadIdx.x;
  if (idx >= S * Dv) return;
  int s = idx / Dv, d = idx % Dv;
  int hh = ss[b * 2 + 0], ww = ss[b * 2 + 1];
  int yy = 0, xx = 0;
  if (s < hh * ww) { yy = s / ww; xx = s % ww; }
  float fy = (yy + 0.5f) * (float)PESv / (float)hh - 0.5f;
  float fx = (xx + 0.5f) * (float)PESv / (float)ww - 0.5f;
  float fy0 = floorf(fy), fx0 = floorf(fx);
  float wy = fy - fy0, wx = fx - fx0;
  int y0 = (int)fy0, x0 = (int)fx0;
  int y0c = min(max(y0, 0), PESv - 1), y1c = min(max(y0 + 1, 0), PESv - 1);
  int x0c = min(max(x0, 0), PESv - 1), x1c = min(max(x0 + 1, 0), PESv - 1);
  float v00 = pe[((size_t)y0c * PESv + x0c) * Dv + d];
  float v01 = pe[((size_t)y0c * PESv + x1c) * Dv + d];
  float v10 = pe[((size_t)y1c * PESv + x0c) * Dv + d];
  float v11 = pe[((size_t)y1c * PESv + x1c) * Dv + d];
  float vv = v00 * (1.f - wy) * (1.f - wx) + v01 * (1.f - wy) * wx +
             v10 * wy * (1.f - wx) + v11 * wy * wx;
  hbuf[((size_t)b * S + s) * Dv + d] += vv;
}

// ---------------------------------------------------------------------------
// Pooling head attention: one block per (head, batch); single query.
// ---------------------------------------------------------------------------
__global__ __launch_bounds__(WG) void head_attn(
    const float* __restrict__ q, const float* __restrict__ k,
    const float* __restrict__ v, const float* __restrict__ mask,
    float* __restrict__ out, int S, int NH, int HDv, float scale) {
  __shared__ float sA[1024];
  __shared__ float red[WG];
  const int b = blockIdx.y, hh = blockIdx.x, tid = threadIdx.x;
  const int D = NH * HDv;
  for (int s = tid; s < S; s += WG) {
    float acc = 0.f;
    for (int d = 0; d < HDv; ++d)
      acc += q[(size_t)b * D + hh * HDv + d] * k[((size_t)b * S + s) * D + hh * HDv + d];
    sA[s] = acc * scale + (1.f - mask[(size_t)b * S + s]) * -3.4028235e38f;
  }
  __syncthreads();
  float mx = -3.4e38f;
  for (int s = tid; s < S; s += WG) mx = fmaxf(mx, sA[s]);
  red[tid] = mx;
  __syncthreads();
  for (int st = WG / 2; st > 0; st >>= 1) {
    if (tid < st) red[tid] = fmaxf(red[tid], red[tid + st]);
    __syncthreads();
  }
  mx = red[0];
  __syncthreads();
  float sum = 0.f;
  for (int s = tid; s < S; s += WG) {
    float p = __expf(sA[s] - mx);
    sA[s] = p;
    sum += p;
  }
  red[tid] = sum;
  __syncthreads();
  for (int st = WG / 2; st > 0; st >>= 1) {
    if (tid < st) red[tid] += red[tid + st];
    __syncthreads();
  }
  float inv = 1.f / red[0];
  for (int d = tid; d < HDv; d += WG) {
    float acc = 0.f;
    for (int s = 0; s < S; ++s)
      acc += sA[s] * v[((size_t)b * S + s) * D + hh * HDv + d];
    out[(size_t)b * D + hh * HDv + d] = acc * inv;
  }
}

// ---------------------------------------------------------------------------
// Tiny GEMM for M=4 head matrices: one thread per output element.
// aStride==0 broadcasts A row 0 (probe). act: tanh-GELU before residual.
// ---------------------------------------------------------------------------
__global__ void small_mm(const float* __restrict__ A, const float* __restrict__ W,
                         const float* __restrict__ bias, const float* __restrict__ resid,
                         float* __restrict__ C, int M, int N, int K, int act, int aStride) {
  int idx = blockIdx.x * blockDim.x + threadIdx.x;
  if (idx >= M * N) return;
  int m = idx / N, n = idx % N;
  float acc = bias ? bias[n] : 0.f;
  const float* ar = A + (size_t)m * aStride;
  for (int kk = 0; kk < K; ++kk) acc += ar[kk] * W[(size_t)kk * N + n];
  if (act)   acc = gelu_tanh(acc);
  if (resid) acc += resid[idx];
  C[idx] = acc;
}

// ---------------------------------------------------------------------------
extern "C" void kernel_launch(void* const* d_in, const int* in_sizes, int n_in,
                              void* d_out, int out_size, void* d_ws, size_t ws_size,
                              hipStream_t stream) {
  (void)in_sizes; (void)n_in; (void)out_size; (void)ws_size;
  const int Bb = 4, Ss = 1024, Dd = 1152, NH = 16, HDv = 72, FFv = 4304,
            Lv = 6, PESv = 16, PIN = 588;
  const float scale = 0.11785113019775793f; // 72^-0.5

  const float* pix     = (const float*)d_in[0];
  const float* amask   = (const float*)d_in[1];
  const int*   ss      = (const int*)d_in[2];
  const float* patch_w = (const float*)d_in[3];
  const float* patch_b = (const float*)d_in[4];
  const float* pos_emb = (const float*)d_in[5];
  const float* ln1w = (const float*)d_in[6];
  const float* ln1b = (const float*)d_in[7];
  const float* qw = (const float*)d_in[8];
  const float* qb = (const float*)d_in[9];
  const float* kw = (const float*)d_in[10];
  const float* kb = (const float*)d_in[11];
  const float* vw = (const float*)d_in[12];
  const float* vb = (const float*)d_in[13];
  const float* ow = (const float*)d_in[14];
  const float* ob = (const float*)d_in[15];
  const float* ln2w = (const float*)d_in[16];
  const float* ln2b = (const float*)d_in[17];
  const float* f1w = (const float*)d_in[18];
  const float* f1b = (const float*)d_in[19];
  const float* f2w = (const float*)d_in[20];
  const float* f2b = (const float*)d_in[21];
  const float* plw = (const float*)d_in[22];
  const float* plb = (const float*)d_in[23];
  const float* probe = (const float*)d_in[24];
  const float* hqw = (const float*)d_in[25];
  const float* hqb = (const float*)d_in[26];
  const float* hkw = (const float*)d_in[27];
  const float* hkb = (const float*)d_in[28];
  const float* hvw = (const float*)d_in[29];
  const float* hvb = (const float*)d_in[30];
  const float* how_ = (const float*)d_in[31];
  const float* hob = (const float*)d_in[32];
  const float* hlnw = (const float*)d_in[33];
  const float* hlnb = (const float*)d_in[34];
  const float* hf1w = (const float*)d_in[35];
  const float* hf1b = (const float*)d_in[36];
  const float* hf2w = (const float*)d_in[37];
  const float* hf2b = (const float*)d_in[38];

  float* last   = (float*)d_out;
  float* pooled = last + (size_t)Bb * Ss * Dd;

  const size_t Hsz = (size_t)Bb * Ss * Dd; // 4,718,592 floats
  float* wsf  = (float*)d_ws;
  float* hbuf = wsf;
  float* xbuf = wsf + Hsz;
  float* qbuf = wsf + 2 * Hsz;
  float* kbuf = wsf + 3 * Hsz;
  float* vbuf = wsf + 4 * Hsz;
  float* abuf = wsf + 5 * Hsz;
  float* ffbuf = qbuf; // reuse q..attn region (4*Hsz >= 4096*4304)
  float* smallb = wsf + 6 * Hsz;
  float* hq   = smallb;
  float* hout = hq + 4 * Dd;
  float* hp   = hout + 4 * Dd;
  float* hpln = hp + 4 * Dd;
  float* hff  = hpln + 4 * Dd;

  const int Mtok = Bb * Ss; // 4096
  dim3 blk(WG);
  dim3 gD((Dd + BN - 1) / BN, (Mtok + BM - 1) / BM);
  dim3 gFF((FFv + BN - 1) / BN, (Mtok + BM - 1) / BM);

  // patch embedding + position embedding
  gemm_bf16<<<gD, blk, 0, stream>>>(pix, patch_w, patch_b, nullptr, hbuf, Mtok, Dd, PIN, 0);
  posemb_add<<<dim3((Ss * Dd + WG - 1) / WG, Bb), blk, 0, stream>>>(hbuf, pos_emb, ss, Ss, Dd, PESv);

  // transformer layers
  for (int l = 0; l < Lv; ++l) {
    ln_fwd<<<Mtok, blk, 0, stream>>>(hbuf, ln1w + (size_t)l * Dd, ln1b + (size_t)l * Dd, xbuf, Dd);
    gemm_bf16<<<gD, blk, 0, stream>>>(xbuf, qw + (size_t)l * Dd * Dd, qb + (size_t)l * Dd, nullptr, qbuf, Mtok, Dd, Dd, 0);
    gemm_bf16<<<gD, blk, 0, stream>>>(xbuf, kw + (size_t)l * Dd * Dd, kb + (size_t)l * Dd, nullptr, kbuf, Mtok, Dd, Dd, 0);
    gemm_bf16<<<gD, blk, 0, stream>>>(xbuf, vw + (size_t)l * Dd * Dd, vb + (size_t)l * Dd, nullptr, vbuf, Mtok, Dd, Dd, 0);
    flash_attn<<<dim3(Ss / TQ, NH, Bb), blk, 0, stream>>>(qbuf, kbuf, vbuf, amask, abuf, Ss, NH, HDv, scale);
    gemm_bf16<<<gD, blk, 0, stream>>>(abuf, ow + (size_t)l * Dd * Dd, ob + (size_t)l * Dd, hbuf, hbuf, Mtok, Dd, Dd, 0);
    ln_fwd<<<Mtok, blk, 0, stream>>>(hbuf, ln2w + (size_t)l * Dd, ln2b + (size_t)l * Dd, xbuf, Dd);
    gemm_bf16<<<gFF, blk, 0, stream>>>(xbuf, f1w + (size_t)l * Dd * FFv, f1b + (size_t)l * FFv, nullptr, ffbuf, Mtok, FFv, Dd, 1);
    gemm_bf16<<<gD, blk, 0, stream>>>(ffbuf, f2w + (size_t)l * FFv * Dd, f2b + (size_t)l * Dd, hbuf, hbuf, Mtok, Dd, FFv, 0);
  }

  // post layernorm -> first output
  ln_fwd<<<Mtok, blk, 0, stream>>>(hbuf, plw, plb, last, Dd);

  // pooling head
  small_mm<<<(4 * Dd + WG - 1) / WG, blk, 0, stream>>>(probe, hqw, hqb, nullptr, hq, Bb, Dd, Dd, 0, 0);
  gemm_bf16<<<gD, blk, 0, stream>>>(last, hkw, hkb, nullptr, kbuf, Mtok, Dd, Dd, 0);
  gemm_bf16<<<gD, blk, 0, stream>>>(last, hvw, hvb, nullptr, vbuf, Mtok, Dd, Dd, 0);
  head_attn<<<dim3(NH, Bb), blk, 0, stream>>>(hq, kbuf, vbuf, amask, hout, Ss, NH, HDv, scale);
  small_mm<<<(4 * Dd + WG - 1) / WG, blk, 0, stream>>>(hout, how_, hob, nullptr, hp, Bb, Dd, Dd, 0, Dd);
  ln_fwd<<<Bb, blk, 0, stream>>>(hp, hlnw, hlnb, hpln, Dd);
  small_mm<<<(4 * FFv + WG - 1) / WG, blk, 0, stream>>>(hpln, hf1w, hf1b, nullptr, hff, Bb, FFv, Dd, 1, Dd);
  small_mm<<<(4 * Dd + WG - 1) / WG, blk, 0, stream>>>(hff, hf2w, hf2b, hp, pooled, Bb, Dd, FFv, 0, FFv);
}
